// Alignment_39462159516189
// MI455X (gfx1250) — matrix operand order repaired and checked by reading, not compiled, not run on hardware
//
#include <hip/hip_runtime.h>

// ---------------------------------------------------------------------------
// Decomposable attention NLI forward, MI455X (gfx1250, wave32, WMMA).
// All GEMMs run on v_wmma_f32_16x16x32_bf16 (f32 accum). f32 inputs/weights
// are converted to bf16 with native v_cvt during LDS staging; bf16 tiles are
// staged with GLOBAL_LOAD_ASYNC_TO_LDS_B128 (ASYNCcnt) when the builtin is
// available. Sum-pool over L is fused into the compare-GEMM epilogue.
// Workspace use: ~75 MB.
// ---------------------------------------------------------------------------

typedef unsigned short u16;
typedef unsigned int   u32;
typedef __attribute__((ext_vector_type(16))) __bf16 v16bf;
typedef __attribute__((ext_vector_type(2)))  __bf16 bf16x2;
typedef __attribute__((ext_vector_type(8)))  float  v8f;
typedef int v4i __attribute__((vector_size(16)));   // matches async-LDS builtin pointee

#define THREADS 256
#define LDK 40   // 32 k-elems + 8 pad (row = 80B, 16B aligned)

#define AS1 __attribute__((address_space(1)))
#define AS3 __attribute__((address_space(3)))

#if __has_builtin(__builtin_amdgcn_global_load_async_to_lds_b128)
#define HAVE_ASYNC 1
#else
#define HAVE_ASYNC 0
#endif

// ---- scalar helpers -------------------------------------------------------
__device__ __forceinline__ u16 f2bf(float f) {          // native v_cvt
    return __builtin_bit_cast(u16, (__bf16)f);
}
__device__ __forceinline__ u32 pack2bf(float x, float y) {  // v_cvt_pk_bf16_f32
    bf16x2 p = { (__bf16)x, (__bf16)y };
    return __builtin_bit_cast(u32, p);
}
__device__ __forceinline__ float bf2f(u16 h) {
    return __uint_as_float(((u32)h) << 16);
}
__device__ __forceinline__ float fast_tanh(float x) {
#if __has_builtin(__builtin_amdgcn_tanhf)
    return __builtin_amdgcn_tanhf(x);                   // v_tanh_f32
#elif __has_builtin(__builtin_amdgcn_tanh_f32)
    return __builtin_amdgcn_tanh_f32(x);
#else
    return tanhf(x);
#endif
}
__device__ __forceinline__ void wait_async() {
#if HAVE_ASYNC
#if __has_builtin(__builtin_amdgcn_s_wait_asynccnt)
    __builtin_amdgcn_s_wait_asynccnt(0);
#else
    asm volatile("s_wait_asynccnt 0x0" ::: "memory");
#endif
#endif
}

// ---- WMMA fragments -------------------------------------------------------
struct Frag {
    union { v16bf v; uint4 q[2]; };
};

// 16-bit A 16x32 layout: lane-half hi: e0..7 = K[hi*8 .. hi*8+7],
// e8..15 = K[16+hi*8 .. 23+hi*8]  -> two 16B chunks per lane row.
__device__ __forceinline__ void load_a(const u16* rowbase, int hi, Frag& f) {
    const uint4* p = (const uint4*)rowbase;
    f.q[0] = p[hi];
    f.q[1] = p[2 + hi];
}
// B 32x16 staged as LDS[n][k]: lane n = base + (lane&15), same k order as A:
// e0..7 = K[hi*8 .. hi*8+7], e8..15 = K[16+hi*8 .. 23+hi*8].
__device__ __forceinline__ void load_b(const u16* rowbase, int hi, Frag& f) {
    const uint4* p = (const uint4*)rowbase;
    f.q[0] = p[hi];
    f.q[1] = p[2 + hi];
}
__device__ __forceinline__ v8f wmma_bf16(const Frag& a, const Frag& b, v8f c) {
    return __builtin_amdgcn_wmma_f32_16x16x32_bf16(
        false, a.v, false, b.v, (short)0, c, false, false);
}

// ---- LDS staging helpers (256-thread cooperative) -------------------------
// A tile: rows x 32, f32 source row-major (ld) -> bf16 LDS [r][k] stride LDK
__device__ __forceinline__ void stage_a_f32(u16* lds, const float* __restrict__ g,
                                            int ld, int rows, int tid) {
    int total = rows * 8;                      // float4 count
    for (int i = tid; i < total; i += THREADS) {
        int r = i >> 3, c = (i & 7) << 2;
        float4 v = *(const float4*)(g + (size_t)r * ld + c);
        u32* d = (u32*)(lds + r * LDK + c);
        d[0] = pack2bf(v.x, v.y);
        d[1] = pack2bf(v.z, v.w);
    }
}
// A tile from bf16 source: async DMA straight into LDS (16B granules)
__device__ __forceinline__ void stage_a_bf16(u16* lds, const u16* __restrict__ g,
                                             int ld, int rows, int tid) {
    int total = rows * 4;                      // uint4 (8 bf16) count
    for (int i = tid; i < total; i += THREADS) {
        int r = i >> 2, c = (i & 3) << 3;
#if HAVE_ASYNC
        __builtin_amdgcn_global_load_async_to_lds_b128(
            (AS1 v4i*)(g + (size_t)r * ld + c),
            (AS3 v4i*)(lds + r * LDK + c), 0, 0);
#else
        *(uint4*)(lds + r * LDK + c) = *(const uint4*)(g + (size_t)r * ld + c);
#endif
    }
}
// B tile: f32 [K][N] row-major, take 32 x 128 at (k0,n0) -> LDS [n][k] (transposed)
__device__ __forceinline__ void stage_b_t_f32(u16* lds, const float* __restrict__ g,
                                              int ld, int tid) {
    for (int i = tid; i < 128 * 32; i += THREADS) {
        int k = i >> 7, n = i & 127;           // consecutive tid -> consecutive n
        lds[n * LDK + k] = f2bf(g[(size_t)k * ld + n]);
    }
}

// ===========================================================================
// K0: zero the aggregate buffer (atomically accumulated later)
// ===========================================================================
__global__ void zero_f32(float* p, int n) {
    int i = blockIdx.x * blockDim.x + threadIdx.x;
    if (i < n) p[i] = 0.0f;
}

// ===========================================================================
// K1: F = tanh(X @ W_F), X = [P;H] (16384 x 1024), W_F 1024x1024 -> bf16 F
// grid (N/128=8, M/128=128), block 256
// ===========================================================================
__global__ __launch_bounds__(THREADS)
void proj_gemm(const float* __restrict__ P, const float* __restrict__ H,
               const float* __restrict__ WF, u16* __restrict__ F) {
    __shared__ u16 As[128 * LDK];
    __shared__ u16 Bs[128 * LDK];
    int tid = threadIdx.x, lane = tid & 31, wave = tid >> 5;
    int hi = lane >> 4, ln = lane & 15;
    int wm = wave >> 1, wn = wave & 1;          // 4x2 waves, wave tile 32x64
    int mB = blockIdx.y * 128, nB = blockIdx.x * 128;
    const float* src = (mB < 8192) ? (P + (size_t)mB * 1024)
                                   : (H + (size_t)(mB - 8192) * 1024);
    v8f acc[2][4] = {};
    for (int k0 = 0; k0 < 1024; k0 += 32) {
        stage_a_f32(As, src + k0, 1024, 128, tid);
        stage_b_t_f32(Bs, WF + (size_t)k0 * 1024 + nB, 1024, tid);
        __syncthreads();
        Frag a[2], b[4];
        for (int i = 0; i < 2; ++i) load_a(As + (wm * 32 + i * 16 + ln) * LDK, hi, a[i]);
        for (int j = 0; j < 4; ++j) load_b(Bs + (wn * 64 + j * 16 + ln) * LDK, hi, b[j]);
        for (int i = 0; i < 2; ++i)
            for (int j = 0; j < 4; ++j)
                acc[i][j] = wmma_bf16(a[i], b[j], acc[i][j]);
        __syncthreads();
    }
    for (int i = 0; i < 2; ++i)
        for (int j = 0; j < 4; ++j)
            for (int v = 0; v < 8; ++v) {
                int row = mB + wm * 32 + i * 16 + v + 8 * hi;
                int col = nB + wn * 64 + j * 16 + ln;
                F[(size_t)row * 1024 + col] = f2bf(fast_tanh(acc[i][j][v]));
            }
}

// ===========================================================================
// K2: e = Fp[b] @ Fh[b]^T (32-row strip x 256), softmax rows, write attn +
// attn^T (bf16). grid (8 strips, 32 batches)
// ===========================================================================
__global__ __launch_bounds__(THREADS)
void attn_softmax(const u16* __restrict__ F, u16* __restrict__ attn,
                  u16* __restrict__ attnT) {
    __shared__ u16 As[32 * LDK];
    __shared__ u16 Bs[256 * LDK];
    __shared__ float sc[32 * 256];
    int tid = threadIdx.x, lane = tid & 31, wave = tid >> 5;
    int hi = lane >> 4, ln = lane & 15;
    int wm = wave >> 2, wn = wave & 3;          // 2x4 waves, wave tile 16x64
    int b = blockIdx.y, strip = blockIdx.x;     // 32 rows each
    const u16* Fp = F + ((size_t)(b * 256 + strip * 32)) * 1024;
    const u16* Fh = F + ((size_t)(8192 + b * 256)) * 1024;
    v8f acc[4] = {};
    for (int k0 = 0; k0 < 1024; k0 += 32) {
        stage_a_bf16(As, Fp + k0, 1024, 32, tid);
        stage_a_bf16(Bs, Fh + k0, 1024, 256, tid);  // [n][k] since B = Fh^T
        wait_async();
        __syncthreads();
        Frag a, bb;
        load_a(As + (wm * 16 + ln) * LDK, hi, a);
        for (int j = 0; j < 4; ++j) {
            load_b(Bs + (wn * 64 + j * 16 + ln) * LDK, hi, bb);
            acc[j] = wmma_bf16(a, bb, acc[j]);
        }
        __syncthreads();
    }
    for (int j = 0; j < 4; ++j)
        for (int v = 0; v < 8; ++v) {
            int row = wm * 16 + v + 8 * hi;
            int col = wn * 64 + j * 16 + ln;
            sc[row * 256 + col] = acc[j][v];
        }
    __syncthreads();
    if (tid < 32) {                              // one row per thread
        int r = tid;
        float mx = -3.0e38f;
        for (int h = 0; h < 256; ++h) mx = fmaxf(mx, sc[r * 256 + h]);
        float s = 0.0f;
        for (int h = 0; h < 256; ++h) {
            float e = __expf(sc[r * 256 + h] - mx);
            sc[r * 256 + h] = e;
            s += e;
        }
        float inv = 1.0f / s;
        int p = strip * 32 + r;
        for (int h = 0; h < 256; ++h) {
            u16 a = f2bf(sc[r * 256 + h] * inv);
            attn[((size_t)(b * 256 + p)) * 256 + h] = a;
            attnT[((size_t)(b * 256 + h)) * 256 + p] = a;
        }
    }
}

// ===========================================================================
// K3: betas = attn @ H ; alphas = attn^T @ P (M=256,N=1024,K=256) -> bf16
// grid (8, 2, 64) z = b*2 + which
// ===========================================================================
__global__ __launch_bounds__(THREADS)
void ba_gemm(const u16* __restrict__ attn, const u16* __restrict__ attnT,
             const float* __restrict__ H, const float* __restrict__ P,
             u16* __restrict__ betas, u16* __restrict__ alphas) {
    __shared__ u16 As[128 * LDK];
    __shared__ u16 Bs[128 * LDK];
    int tid = threadIdx.x, lane = tid & 31, wave = tid >> 5;
    int hi = lane >> 4, ln = lane & 15;
    int wm = wave >> 1, wn = wave & 1;
    int which = blockIdx.z & 1, b = blockIdx.z >> 1;
    int mB = blockIdx.y * 128, nB = blockIdx.x * 128;
    const u16* A = (which == 0 ? attn : attnT) + (size_t)b * 256 * 256;
    const float* Bsrc = (which == 0 ? H : P) + (size_t)b * 256 * 1024;
    u16* out = (which == 0 ? betas : alphas) + (size_t)b * 256 * 1024;
    v8f acc[2][4] = {};
    for (int k0 = 0; k0 < 256; k0 += 32) {
        stage_a_bf16(As, A + (size_t)mB * 256 + k0, 256, 128, tid);
        stage_b_t_f32(Bs, Bsrc + (size_t)k0 * 1024 + nB, 1024, tid);
        wait_async();
        __syncthreads();
        Frag a[2], bb[4];
        for (int i = 0; i < 2; ++i) load_a(As + (wm * 32 + i * 16 + ln) * LDK, hi, a[i]);
        for (int j = 0; j < 4; ++j) load_b(Bs + (wn * 64 + j * 16 + ln) * LDK, hi, bb[j]);
        for (int i = 0; i < 2; ++i)
            for (int j = 0; j < 4; ++j)
                acc[i][j] = wmma_bf16(a[i], bb[j], acc[i][j]);
        __syncthreads();
    }
    for (int i = 0; i < 2; ++i)
        for (int j = 0; j < 4; ++j)
            for (int v = 0; v < 8; ++v) {
                int row = mB + wm * 32 + i * 16 + v + 8 * hi;
                int col = nB + wn * 64 + j * 16 + ln;
                out[(size_t)row * 1024 + col] = f2bf(acc[i][j][v]);
            }
}

// ===========================================================================
// K4: V = tanh(concat(X, Y) @ W_G); fused column-sum over rows into agg.
// M=256, N=2048, K=2048 per (b, which). grid (16, 2, 64)
// agg layout: [32][4096] : which=0 -> cols 0..2047, which=1 -> 2048..4095
// ===========================================================================
__global__ __launch_bounds__(THREADS)
void compare_gemm(const float* __restrict__ P, const float* __restrict__ H,
                  const u16* __restrict__ betas, const u16* __restrict__ alphas,
                  const float* __restrict__ WG, float* __restrict__ agg) {
    __shared__ u16 As[128 * LDK];
    __shared__ u16 Bs[128 * LDK];
    __shared__ float colsum[128];
    int tid = threadIdx.x, lane = tid & 31, wave = tid >> 5;
    int hi = lane >> 4, ln = lane & 15;
    int wm = wave >> 1, wn = wave & 1;
    int which = blockIdx.z & 1, b = blockIdx.z >> 1;
    int mB = blockIdx.y * 128, nB = blockIdx.x * 128;
    const float* f32src = (which == 0 ? P : H) + (size_t)(b * 256 + mB) * 1024;
    const u16*  bfsrc = (which == 0 ? betas : alphas) + (size_t)(b * 256 + mB) * 1024;
    v8f acc[2][4] = {};
    for (int k0 = 0; k0 < 2048; k0 += 32) {
        if (k0 < 1024) stage_a_f32(As, f32src + k0, 1024, 128, tid);
        else           stage_a_bf16(As, bfsrc + (k0 - 1024), 1024, 128, tid);
        stage_b_t_f32(Bs, WG + (size_t)k0 * 2048 + nB, 2048, tid);
        wait_async();
        __syncthreads();
        Frag a[2], bb[4];
        for (int i = 0; i < 2; ++i) load_a(As + (wm * 32 + i * 16 + ln) * LDK, hi, a[i]);
        for (int j = 0; j < 4; ++j) load_b(Bs + (wn * 64 + j * 16 + ln) * LDK, hi, bb[j]);
        for (int i = 0; i < 2; ++i)
            for (int j = 0; j < 4; ++j)
                acc[i][j] = wmma_bf16(a[i], bb[j], acc[i][j]);
        __syncthreads();
    }
    if (tid < 128) colsum[tid] = 0.0f;
    __syncthreads();
    for (int i = 0; i < 2; ++i)
        for (int j = 0; j < 4; ++j) {
            float s = 0.0f;
            for (int v = 0; v < 8; ++v) s += fast_tanh(acc[i][j][v]);
            atomicAdd(&colsum[wn * 64 + j * 16 + ln], s);  // ds_add_f32
        }
    __syncthreads();
    if (tid < 128)
        atomicAdd(&agg[(size_t)b * 4096 + which * 2048 + nB + tid], colsum[tid]);
}

// ===========================================================================
// K5: small-M MLP GEMM: out = tanh(A @ W + bias), M=32, N=2048, K in {4096,2048}
// A = f32 (agg) or bf16 (a1). grid (N/128), block 256 (8 waves, wave 32x16)
// ===========================================================================
__global__ __launch_bounds__(THREADS)
void mlp_gemm(const float* __restrict__ Af32, const u16* __restrict__ Abf,
              const float* __restrict__ W, const float* __restrict__ bias,
              u16* __restrict__ out, int K, int N) {
    __shared__ u16 As[32 * LDK];
    __shared__ u16 Bs[128 * LDK];
    int tid = threadIdx.x, lane = tid & 31, wave = tid >> 5;
    int hi = lane >> 4, ln = lane & 15;
    int nB = blockIdx.x * 128;
    v8f acc[2] = {};
    for (int k0 = 0; k0 < K; k0 += 32) {
        if (Af32) stage_a_f32(As, Af32 + k0, K, 32, tid);
        else      stage_a_bf16(As, Abf + k0, K, 32, tid);
        stage_b_t_f32(Bs, W + (size_t)k0 * N + nB, N, tid);
        wait_async();
        __syncthreads();
        Frag a[2], bb;
        for (int i = 0; i < 2; ++i) load_a(As + (i * 16 + ln) * LDK, hi, a[i]);
        load_b(Bs + (wave * 16 + ln) * LDK, hi, bb);
        for (int i = 0; i < 2; ++i) acc[i] = wmma_bf16(a[i], bb, acc[i]);
        __syncthreads();
    }
    int col = nB + wave * 16 + ln;
    float bv = bias[col];
    for (int i = 0; i < 2; ++i)
        for (int v = 0; v < 8; ++v) {
            int row = i * 16 + v + 8 * hi;
            out[(size_t)row * N + col] = f2bf(fast_tanh(acc[i][v] + bv));
        }
}

// ===========================================================================
// K6: logits = a2 @ W3 + b3   ([32,2048] @ [2048,3]) -> f32 out
// ===========================================================================
__global__ __launch_bounds__(THREADS)
void logits_kernel(const u16* __restrict__ a2, const float* __restrict__ W3,
                   const float* __restrict__ b3, float* __restrict__ out) {
    __shared__ float red[THREADS];
    int b = blockIdx.y, c = blockIdx.x;
    float s = 0.0f;
    for (int k = threadIdx.x; k < 2048; k += THREADS)
        s += bf2f(a2[(size_t)b * 2048 + k]) * W3[(size_t)k * 3 + c];
    red[threadIdx.x] = s;
    __syncthreads();
    for (int off = THREADS / 2; off > 0; off >>= 1) {
        if (threadIdx.x < off) red[threadIdx.x] += red[threadIdx.x + off];
        __syncthreads();
    }
    if (threadIdx.x == 0) out[b * 3 + c] = red[0] + b3[c];
}

// ===========================================================================
extern "C" void kernel_launch(void* const* d_in, const int* in_sizes, int n_in,
                              void* d_out, int out_size, void* d_ws, size_t ws_size,
                              hipStream_t stream) {
    const float* prem = (const float*)d_in[0];
    const float* hyp  = (const float*)d_in[1];
    const float* WF   = (const float*)d_in[2];
    const float* WG   = (const float*)d_in[3];
    const float* W1   = (const float*)d_in[4];
    const float* b1   = (const float*)d_in[5];
    const float* W2   = (const float*)d_in[6];
    const float* b2   = (const float*)d_in[7];
    const float* W3   = (const float*)d_in[8];
    const float* b3   = (const float*)d_in[9];
    float* out = (float*)d_out;

    // workspace carve-out (~75 MB)
    char* ws = (char*)d_ws;
    u16* F      = (u16*)ws;  ws += (size_t)16384 * 1024 * 2;      // proj (P;H) bf16
    u16* attn   = (u16*)ws;  ws += (size_t)32 * 256 * 256 * 2;
    u16* attnT  = (u16*)ws;  ws += (size_t)32 * 256 * 256 * 2;
    u16* betas  = (u16*)ws;  ws += (size_t)32 * 256 * 1024 * 2;
    u16* alphas = (u16*)ws;  ws += (size_t)32 * 256 * 1024 * 2;
    float* agg  = (float*)ws; ws += (size_t)32 * 4096 * 4;
    u16* a1     = (u16*)ws;  ws += (size_t)32 * 2048 * 2;
    u16* a2     = (u16*)ws;  ws += (size_t)32 * 2048 * 2;

    zero_f32<<<(32 * 4096 + 255) / 256, 256, 0, stream>>>(agg, 32 * 4096);
    proj_gemm<<<dim3(8, 128), THREADS, 0, stream>>>(prem, hyp, WF, F);
    attn_softmax<<<dim3(8, 32), THREADS, 0, stream>>>(F, attn, attnT);
    ba_gemm<<<dim3(8, 2, 64), THREADS, 0, stream>>>(attn, attnT, hyp, prem, betas, alphas);
    compare_gemm<<<dim3(16, 2, 64), THREADS, 0, stream>>>(prem, hyp, betas, alphas, WG, agg);
    mlp_gemm<<<dim3(16), THREADS, 0, stream>>>(agg, nullptr, W1, b1, a1, 4096, 2048);
    mlp_gemm<<<dim3(16), THREADS, 0, stream>>>(nullptr, a1, W2, b2, a2, 2048, 2048);
    logits_kernel<<<dim3(3, 32), THREADS, 0, stream>>>(a2, W3, b3, out);
}
